// Decoder_49340584296828
// MI455X (gfx1250) — matrix operand order repaired and checked by reading, not compiled, not run on hardware
//
#include <hip/hip_runtime.h>

#define Bn 64
#define Tn 400
#define Hn 256
#define En 128
#define Vn 50000
#define Xn 500
#define VXn 50500

typedef __attribute__((ext_vector_type(16))) __bf16 v16bf;
typedef __attribute__((ext_vector_type(8)))  __bf16 v8bf;
typedef __attribute__((ext_vector_type(8)))  float  v8f;

__device__ __forceinline__ float sigmoidf_(float x) { return 1.0f / (1.0f + __expf(-x)); }

// ---------------- x = [c_t_1, emb[y]] @ Wxc^T + bxc ----------------
__global__ void k_x(const float* __restrict__ c_t_1, const int* __restrict__ y,
                    const float* __restrict__ emb, const float* __restrict__ Wxc,
                    const float* __restrict__ bxc, float* __restrict__ x) {
    int idx = blockIdx.x * blockDim.x + threadIdx.x;   // 64*128
    int e = idx >> 6, b = idx & 63;
    const float* w = Wxc + e * 640;
    const float* ct = c_t_1 + b * 512;
    float acc = bxc[e];
    #pragma unroll 4
    for (int k = 0; k < 512; ++k) acc += ct[k] * w[k];
    const float* em = emb + (size_t)y[b] * En;
    #pragma unroll 4
    for (int k = 0; k < 128; ++k) acc += em[k] * w[512 + k];
    x[b * En + e] = acc;
}

// ---------------- gates = x @ W_ih^T + h0 @ W_hh^T + b ----------------
__global__ void k_gates(const float* __restrict__ x, const float* __restrict__ h0,
                        const float* __restrict__ W_ih, const float* __restrict__ W_hh,
                        const float* __restrict__ b_ih, const float* __restrict__ b_hh,
                        float* __restrict__ gates) {
    int idx = blockIdx.x * blockDim.x + threadIdx.x;   // 64*1024
    int j = idx >> 6, b = idx & 63;
    float acc = b_ih[j] + b_hh[j];
    const float* wi = W_ih + j * 128;
    const float* xb = x + b * 128;
    #pragma unroll 4
    for (int k = 0; k < 128; ++k) acc += xb[k] * wi[k];
    const float* wh = W_hh + j * 256;
    const float* hb = h0 + b * 256;
    #pragma unroll 4
    for (int k = 0; k < 256; ++k) acc += hb[k] * wh[k];
    gates[b * 1024 + j] = acc;
}

// ---------------- LSTM cell elementwise ----------------
__global__ void k_cell(const float* __restrict__ gates, const float* __restrict__ c0,
                       float* __restrict__ hs, float* __restrict__ cs) {
    int idx = blockIdx.x * blockDim.x + threadIdx.x;   // 64*256
    int b = idx >> 8, h = idx & 255;
    const float* g = gates + b * 1024;
    float gi = g[h], gf = g[256 + h], gg = g[512 + h], go = g[768 + h];
    float c = sigmoidf_(gf) * c0[b * 256 + h] + sigmoidf_(gi) * tanhf(gg);
    float hh = sigmoidf_(go) * tanhf(c);
    cs[b * 256 + h] = c;
    hs[b * 256 + h] = hh;
}

// ---------------- dec_fea = [hs, cs] @ Wdp^T + bdp ----------------
__global__ void k_decfea(const float* __restrict__ hs, const float* __restrict__ cs,
                         const float* __restrict__ Wdp, const float* __restrict__ bdp,
                         float* __restrict__ decfea) {
    int idx = blockIdx.x * blockDim.x + threadIdx.x;   // 64*512
    int n = idx >> 6, b = idx & 63;
    const float* w = Wdp + n * 512;
    float acc = bdp[n];
    const float* h = hs + b * 256;
    const float* c = cs + b * 256;
    #pragma unroll 4
    for (int k = 0; k < 256; ++k) acc += h[k] * w[k];
    #pragma unroll 4
    for (int k = 0; k < 256; ++k) acc += c[k] * w[256 + k];
    decfea[b * 512 + n] = acc;
}

// ---------------- scores[b,t] = Wv . tanh(enc_feature + dec_fea) (wave per (b,t)) ----------------
__global__ void k_scores(const float* __restrict__ ef, const float* __restrict__ decfea,
                         const float* __restrict__ Wv, float* __restrict__ scores) {
    int gid = blockIdx.x * (blockDim.x >> 5) + (threadIdx.x >> 5);
    int lane = threadIdx.x & 31;
    int b = gid / Tn, t = gid - b * Tn;
    const float4* e4 = (const float4*)(ef + ((size_t)(b * Tn + t)) * 512 + lane * 16);
    const float4* d4 = (const float4*)(decfea + b * 512 + lane * 16);
    const float4* w4 = (const float4*)(Wv + lane * 16);
    float acc = 0.f;
    #pragma unroll
    for (int i = 0; i < 4; ++i) {
        float4 a = e4[i], d = d4[i], w = w4[i];
        acc += w.x * tanhf(a.x + d.x) + w.y * tanhf(a.y + d.y) +
               w.z * tanhf(a.z + d.z) + w.w * tanhf(a.w + d.w);
    }
    #pragma unroll
    for (int off = 16; off > 0; off >>= 1) acc += __shfl_down(acc, off, 32);
    if (lane == 0) scores[b * Tn + t] = acc;
}

// ---------------- softmax over T, mask, renorm, + stmt ----------------
__global__ void k_attn(const float* __restrict__ scores, const float* __restrict__ mask,
                       const float* __restrict__ stmt, float* __restrict__ attn) {
    int b = blockIdx.x, tid = threadIdx.x;             // 128 threads
    __shared__ float red[128];
    const float* s = scores + b * Tn;
    float m = -1e30f;
    for (int t = tid; t < Tn; t += 128) m = fmaxf(m, s[t]);
    red[tid] = m; __syncthreads();
    for (int k = 64; k > 0; k >>= 1) { if (tid < k) red[tid] = fmaxf(red[tid], red[tid + k]); __syncthreads(); }
    m = red[0]; __syncthreads();
    float sum = 0.f;
    for (int t = tid; t < Tn; t += 128) sum += __expf(s[t] - m);
    red[tid] = sum; __syncthreads();
    for (int k = 64; k > 0; k >>= 1) { if (tid < k) red[tid] += red[tid + k]; __syncthreads(); }
    float denom = red[0]; __syncthreads();
    float s2 = 0.f;
    for (int t = tid; t < Tn; t += 128) s2 += __expf(s[t] - m) / denom * mask[b * Tn + t];
    red[tid] = s2; __syncthreads();
    for (int k = 64; k > 0; k >>= 1) { if (tid < k) red[tid] += red[tid + k]; __syncthreads(); }
    float norm = red[0];
    for (int t = tid; t < Tn; t += 128) {
        float a = __expf(s[t] - m) / denom * mask[b * Tn + t];
        attn[b * Tn + t] = a / norm + stmt[b * Tn + t] * mask[b * Tn + t];
    }
}

// ---------------- c_t[b,n] = sum_t attn[b,t] * enc_out[b,t,n] ----------------
__global__ void k_ctx(const float* __restrict__ attn, const float* __restrict__ enc_out,
                      float* __restrict__ ct) {
    int b = blockIdx.x, n = threadIdx.x;               // 512 threads
    const float* eb = enc_out + (size_t)b * Tn * 512;
    const float* ab = attn + b * Tn;
    float acc = 0.f;
    for (int t = 0; t < Tn; ++t) acc += ab[t] * eb[(size_t)t * 512 + n];
    ct[b * 512 + n] = acc;
}

// ---------------- p_gen (wave per b) ----------------
__global__ void k_pgen(const float* __restrict__ ct, const float* __restrict__ hs,
                       const float* __restrict__ cs, const float* __restrict__ x,
                       const float* __restrict__ Wpg, const float* __restrict__ bpg,
                       float* __restrict__ pg) {
    int b = blockIdx.x, lane = threadIdx.x;
    float acc = 0.f;
    for (int k = lane; k < 1152; k += 32) {
        float v;
        if (k < 512) v = ct[b * 512 + k];
        else if (k < 768) v = hs[b * 256 + (k - 512)];
        else if (k < 1024) v = cs[b * 256 + (k - 768)];
        else v = x[b * 128 + (k - 1024)];
        acc += Wpg[k] * v;
    }
    #pragma unroll
    for (int off = 16; off > 0; off >>= 1) acc += __shfl_down(acc, off, 32);
    if (lane == 0) pg[b] = sigmoidf_(acc + bpg[0]);
}

// ---------------- output = [hs, ct] @ Wo1^T + bo1, stored as bf16 A-matrix ----------------
__global__ void k_out1(const float* __restrict__ hs, const float* __restrict__ ct,
                       const float* __restrict__ Wo1, const float* __restrict__ bo1,
                       __bf16* __restrict__ Abf) {
    int idx = blockIdx.x * blockDim.x + threadIdx.x;   // 64*256
    int n = idx >> 6, b = idx & 63;
    const float* w = Wo1 + n * 768;
    float acc = bo1[n];
    const float* h = hs + b * 256;
    const float* c = ct + b * 512;
    #pragma unroll 4
    for (int k = 0; k < 256; ++k) acc += h[k] * w[k];
    #pragma unroll 4
    for (int k = 0; k < 512; ++k) acc += c[k] * w[256 + k];
    Abf[b * 256 + n] = (__bf16)acc;
}

// ---------------- logits GEMM via bf16 WMMA: [64,256] x [256,50000] ----------------
// One wave -> 64(M) x 16(N) tile; B (Wo2 rows) read once device-wide.
// K loop fully unrolled: 32 static v_wmma, whole dependency graph exposed to scheduler.
__global__ void k_logits(const __bf16* __restrict__ Abf, const float* __restrict__ Wo2,
                         const float* __restrict__ bo2, float* __restrict__ out /*VXn stride*/) {
    int tile = blockIdx.x * (blockDim.x >> 5) + (threadIdx.x >> 5);
    if (tile >= Vn / 16) return;                       // wave-uniform guard: EXEC stays all-ones
    int lane = threadIdx.x & 31;
    int n = lane & 15;
    int khalf = lane >> 4;                             // 0: lanes 0-15, 1: lanes 16-31
    int vrow = tile * 16 + n;
    const float* wrow = Wo2 + (size_t)vrow * 256;

    v8f acc[4];
    #pragma unroll
    for (int mt = 0; mt < 4; ++mt)
        #pragma unroll
        for (int r = 0; r < 8; ++r) acc[mt][r] = 0.f;

    #pragma unroll
    for (int kb = 0; kb < 8; ++kb) {
        // B fragment: 16 contiguous K at k0, column n
        int k0 = kb * 32 + khalf * 16;
        v16bf bfrag;
        const float4* w4 = (const float4*)(wrow + k0);
        #pragma unroll
        for (int i = 0; i < 4; ++i) {
            float4 f = w4[i];
            bfrag[4 * i + 0] = (__bf16)f.x;
            bfrag[4 * i + 1] = (__bf16)f.y;
            bfrag[4 * i + 2] = (__bf16)f.z;
            bfrag[4 * i + 3] = (__bf16)f.w;
        }
        // A fragments: row M = mt*16 + n, K runs [ak0, ak0+16) contiguous
        int ak0 = kb * 32 + khalf * 16;
        #pragma unroll
        for (int mt = 0; mt < 4; ++mt) {
            const __bf16* ar = Abf + (size_t)(mt * 16 + n) * 256;
            v8bf a0 = *(const v8bf*)(ar + ak0);
            v8bf a1 = *(const v8bf*)(ar + ak0 + 8);
            v16bf af;
            #pragma unroll
            for (int i = 0; i < 8; ++i) { af[i] = a0[i]; af[i + 8] = a1[i]; }
            acc[mt] = __builtin_amdgcn_wmma_f32_16x16x32_bf16(
                false, af, false, bfrag, (short)0, acc[mt], false, false);
        }
    }
    float bias = bo2[vrow];
    #pragma unroll
    for (int mt = 0; mt < 4; ++mt)
        #pragma unroll
        for (int r = 0; r < 8; ++r) {
            int m = mt * 16 + khalf * 8 + r;
            out[(size_t)m * VXn + vrow] = acc[mt][r] + bias;
        }
}

// ---------------- vocab softmax in place (online max+sum), scale by p_gen, extras ----------------
__global__ void k_vsm(float* __restrict__ out, const float* __restrict__ pg,
                      const float* __restrict__ extraz) {
    int b = blockIdx.x, tid = threadIdx.x;             // 512 threads
    float* row = out + (size_t)b * VXn;
    __shared__ float redm[512];
    __shared__ float reds[512];
    // single online pass: running (max, scaled-sum)
    float m = -1e30f, s = 0.f;
    for (int v = tid; v < Vn; v += 512) {
        float x = row[v];
        float nm = fmaxf(m, x);
        s = s * __expf(m - nm) + __expf(x - nm);
        m = nm;
    }
    redm[tid] = m; reds[tid] = s; __syncthreads();
    for (int k = 256; k > 0; k >>= 1) {
        if (tid < k) {
            float m1 = redm[tid], m2 = redm[tid + k];
            float M = fmaxf(m1, m2);
            reds[tid] = reds[tid] * __expf(m1 - M) + reds[tid + k] * __expf(m2 - M);
            redm[tid] = M;
        }
        __syncthreads();
    }
    m = redm[0];
    float inv = pg[b] / reds[0];
    for (int v = tid; v < Vn; v += 512) row[v] = __expf(row[v] - m) * inv;
    for (int v = Vn + tid; v < VXn; v += 512) row[v] = extraz[b * Xn + (v - Vn)];
}

// ---------------- scatter-add (1-p_gen)*attn into final dist ----------------
__global__ void k_scatter(float* __restrict__ out, const float* __restrict__ attn,
                          const float* __restrict__ pg, const int* __restrict__ idx) {
    int i = blockIdx.x * blockDim.x + threadIdx.x;     // 64*400
    int b = i / Tn, t = i - b * Tn;
    float v = (1.0f - pg[b]) * attn[b * Tn + t];
    atomicAdd(&out[(size_t)b * VXn + idx[b * Tn + t]], v);
}

// ---------------- passthrough coverage ----------------
__global__ void k_copy(const float* __restrict__ src, float* __restrict__ dst, int n) {
    int i = blockIdx.x * blockDim.x + threadIdx.x;
    if (i < n) dst[i] = src[i];
}

extern "C" void kernel_launch(void* const* d_in, const int* in_sizes, int n_in,
                              void* d_out, int out_size, void* d_ws, size_t ws_size,
                              hipStream_t stream) {
    const int*   y      = (const int*)d_in[0];
    const float* h0     = (const float*)d_in[1];
    const float* c0     = (const float*)d_in[2];
    const float* encout = (const float*)d_in[3];
    const float* encf   = (const float*)d_in[4];
    const float* stmt   = (const float*)d_in[5];
    const float* mask   = (const float*)d_in[6];
    const float* c_t_1  = (const float*)d_in[7];
    const float* extraz = (const float*)d_in[8];
    const int*   extv   = (const int*)d_in[9];
    const float* cov    = (const float*)d_in[10];
    // d_in[11] = step (unused)
    const float* emb    = (const float*)d_in[12];
    const float* Wxc    = (const float*)d_in[13];
    const float* bxc    = (const float*)d_in[14];
    const float* W_ih   = (const float*)d_in[15];
    const float* W_hh   = (const float*)d_in[16];
    const float* b_ih   = (const float*)d_in[17];
    const float* b_hh   = (const float*)d_in[18];
    const float* Wdp    = (const float*)d_in[19];
    const float* bdp    = (const float*)d_in[20];
    const float* Wv     = (const float*)d_in[21];
    const float* Wpg    = (const float*)d_in[22];
    const float* bpg    = (const float*)d_in[23];
    const float* Wo1    = (const float*)d_in[24];
    const float* bo1    = (const float*)d_in[25];
    const float* Wo2    = (const float*)d_in[26];
    const float* bo2    = (const float*)d_in[27];

    // workspace layout (floats)
    float* ws      = (float*)d_ws;
    float* w_x     = ws;                  //  8192
    float* w_gates = ws + 8192;           // 65536
    float* w_dec   = ws + 73728;          // 32768
    float* w_scr   = ws + 106496;         // 25600
    __bf16* w_abf  = (__bf16*)(ws + 132096); // 64*256 bf16

    // d_out layout (return-order concat)
    float* out    = (float*)d_out;
    float* o_fin  = out;                  // 64*50500
    float* o_hs   = out + 3232000;        // 64*256
    float* o_cs   = o_hs + 16384;         // 64*256
    float* o_ct   = o_cs + 16384;         // 64*512
    float* o_attn = o_ct + 32768;         // 64*400
    float* o_pg   = o_attn + 25600;       // 64
    float* o_cov  = o_pg + 64;            // 64*400

    k_x      <<<32,  256, 0, stream>>>(c_t_1, y, emb, Wxc, bxc, w_x);
    k_gates  <<<256, 256, 0, stream>>>(w_x, h0, W_ih, W_hh, b_ih, b_hh, w_gates);
    k_cell   <<<64,  256, 0, stream>>>(w_gates, c0, o_hs, o_cs);
    k_decfea <<<128, 256, 0, stream>>>(o_hs, o_cs, Wdp, bdp, w_dec);
    k_scores <<<3200, 256, 0, stream>>>(encf, w_dec, Wv, w_scr);
    k_attn   <<<64,  128, 0, stream>>>(w_scr, mask, stmt, o_attn);
    k_ctx    <<<64,  512, 0, stream>>>(o_attn, encout, o_ct);
    k_pgen   <<<64,  32,  0, stream>>>(o_ct, o_hs, o_cs, w_x, Wpg, bpg, o_pg);
    k_out1   <<<64,  256, 0, stream>>>(o_hs, o_ct, Wo1, bo1, w_abf);
    k_logits <<<782, 128, 0, stream>>>(w_abf, Wo2, bo2, o_fin);
    k_vsm    <<<64,  512, 0, stream>>>(o_fin, o_pg, extraz);
    k_scatter<<<100, 256, 0, stream>>>(o_fin, o_attn, o_pg, extv);
    k_copy   <<<100, 256, 0, stream>>>(cov, o_cov, Bn * Tn);
}
